// CoarsenBlock_19507741458895
// MI455X (gfx1250) — matrix operand order repaired and checked by reading, not compiled, not run on hardware
//
#include <hip/hip_runtime.h>
#include <math.h>

#define BB 16
#define NN 1024
#define CC 128

typedef __attribute__((ext_vector_type(16))) __bf16 v16bf;
typedef __attribute__((ext_vector_type(8)))  float  v8f;

__device__ __forceinline__ float wave_sum(float v) {
#pragma unroll
    for (int off = 16; off > 0; off >>= 1)
        v += __shfl_xor(v, off, 32);
    return v;
}

// ---------------------------------------------------------------------------
// Kernel 1: per-row stats.  xw = x@weight, deg for gcn (diag set to 1),
// deg for norm_adj (adj + I), mask = rowsum(adj) > 0.  One wave per row.
// ---------------------------------------------------------------------------
__global__ void k_rowstats(const float* __restrict__ x,
                           const float* __restrict__ adj,
                           const float* __restrict__ weight,
                           float* __restrict__ xw,
                           float* __restrict__ deg_gcn,
                           float* __restrict__ degA,
                           float* __restrict__ maskv) {
    int wid  = (blockIdx.x * blockDim.x + threadIdx.x) >> 5;
    int lane = threadIdx.x & 31;
    if (wid >= BB * NN) return;
    int b = wid >> 10, n = wid & (NN - 1);

    const float* arow = adj + ((size_t)b * NN + n) * NN;
    float s = 0.f;
    for (int m = lane; m < NN; m += 32) s += arow[m];
    s = wave_sum(s);

    const float* xrow = x + ((size_t)b * NN + n) * CC;
    float t = 0.f;
    for (int c = lane; c < CC; c += 32) t += xrow[c] * weight[c];
    t = wave_sum(t);

    if (lane == 0) {
        float diag = arow[n];
        float dg = fmaxf(s - diag + 1.0f, 1.0f);  // diag set to 1
        float da = fmaxf(s + 1.0f, 1.0f);         // adj + I
        xw[wid]      = t;
        deg_gcn[wid] = rsqrtf(dg);
        degA[wid]    = rsqrtf(da);
        maskv[wid]   = (s > 0.f) ? 1.f : 0.f;
    }
}

// ---------------------------------------------------------------------------
// Kernel 2: gcn matvec + alpha = sigmoid(gcn^2).  One wave per row.
// ---------------------------------------------------------------------------
__global__ void k_gcn_alpha(const float* __restrict__ adj,
                            const float* __restrict__ xw,
                            const float* __restrict__ deg_gcn,
                            const float* __restrict__ bias,
                            float* __restrict__ alpha) {
    int wid  = (blockIdx.x * blockDim.x + threadIdx.x) >> 5;
    int lane = threadIdx.x & 31;
    if (wid >= BB * NN) return;
    int b = wid >> 10, n = wid & (NN - 1);

    const float* arow = adj + ((size_t)b * NN + n) * NN;
    const float* dg  = deg_gcn + b * NN;
    const float* xwb = xw + b * NN;

    float s = 0.f;
    for (int m = lane; m < NN; m += 32) s += arow[m] * dg[m] * xwb[m];
    s = wave_sum(s);

    if (lane == 0) {
        float yn = dg[n] * xwb[n];
        float sc = s - arow[n] * yn + yn;   // replace diag term with 1*y[n]
        float g  = dg[n] * sc + bias[0];
        float a2 = g * g;
        alpha[wid] = 1.f / (1.f + expf(-a2));
    }
}

// ---------------------------------------------------------------------------
// Kernel 3: per-batch k-th largest of alpha via rank counting (N=1024).
// ---------------------------------------------------------------------------
__global__ void k_select(const float* __restrict__ alpha,
                         const int* __restrict__ bnn,
                         float* __restrict__ cut) {
    __shared__ float sa[NN];
    int b = blockIdx.x, t = threadIdx.x;
    sa[t] = alpha[b * NN + t];
    __syncthreads();

    int n = bnn[b];
    if (n <= 1) { if (t == 0) cut[b] = 0.f; return; }
    int k = (n + 1) / 2 + 1;            // ceil(n*0.5)+1

    float v = sa[t];
    int cg = 0, cge = 0;
    for (int j = 0; j < NN; ++j) {
        float u = sa[j];
        cg  += (u > v)  ? 1 : 0;
        cge += (u >= v) ? 1 : 0;
    }
    if (cg <= k - 1 && k - 1 < cge) cut[b] = v;
}

// ---------------------------------------------------------------------------
// Kernel 4: S[n,m] = mask[n]*dA[n]*(adj[n,m]+[m==n])*dA[m]*relu(alpha[m]-cut)
//           then L1-normalize each row.  Writes f32 S (output) and bf16 S
//           (GEMM operand).  One wave per row, two passes.
// ---------------------------------------------------------------------------
__global__ void k_build_s(const float* __restrict__ adj,
                          const float* __restrict__ alpha,
                          const float* __restrict__ degA,
                          const float* __restrict__ maskv,
                          const float* __restrict__ cut,
                          float* __restrict__ S,
                          __bf16* __restrict__ S16) {
    int wid  = (blockIdx.x * blockDim.x + threadIdx.x) >> 5;
    int lane = threadIdx.x & 31;
    if (wid >= BB * NN) return;
    int b = wid >> 10, n = wid & (NN - 1);

    const float* arow = adj + ((size_t)b * NN + n) * NN;
    const float* dA = degA + b * NN;
    const float* al = alpha + b * NN;
    float cutb = cut[b];
    float coef = maskv[wid] * dA[n];

    float s = 0.f;
    for (int m = lane; m < NN; m += 32) {
        float A  = arow[m] + ((m == n) ? 1.f : 0.f);
        float ca = fmaxf(al[m] - cutb, 0.f);
        s += fabsf(coef * dA[m] * A * ca);
    }
    s = wave_sum(s);
    float inv = 1.f / fmaxf(s, 1e-12f);

    float*  srow  = S   + ((size_t)b * NN + n) * NN;
    __bf16* s16row = S16 + ((size_t)b * NN + n) * NN;
    for (int m = lane; m < NN; m += 32) {
        float A  = arow[m] + ((m == n) ? 1.f : 0.f);
        float ca = fmaxf(al[m] - cutb, 0.f);
        float val = coef * dA[m] * A * ca * inv;
        srow[m]   = val;
        s16row[m] = (__bf16)val;
    }
}

// ---------------------------------------------------------------------------
// Elementwise f32 -> bf16 conversion (for adj, x GEMM operands).
// ---------------------------------------------------------------------------
__global__ void k_cvt_bf16(const float* __restrict__ in,
                           __bf16* __restrict__ out, long long count) {
    long long i      = (long long)blockIdx.x * blockDim.x + threadIdx.x;
    long long stride = (long long)gridDim.x * blockDim.x;
    for (; i < count; i += stride) out[i] = (__bf16)in[i];
}

// ---------------------------------------------------------------------------
// Tiled bf16 WMMA GEMM:  C[b] = op(A[b]) @ B[b],  bf16 operands already in
// memory, f32 accumulate via v_wmma_f32_16x16x32_bf16, output f32 or bf16.
// Block = 256 threads (8 waves), 128x128 output tile, wave tile = 32x64.
// Tile staging copies packed pairs (32-bit) of bf16; b-fragments are double
// buffered so ds_load of fragment j+1 overlaps the WMMA pair of fragment j.
// ---------------------------------------------------------------------------
__global__ void __launch_bounds__(256, 2)
k_wmma_gemm(const __bf16* __restrict__ A, long long ldA, long long sA,
            const __bf16* __restrict__ B, long long ldB, long long sB,
            float* __restrict__ Cf, __bf16* __restrict__ Cb,
            long long ldC, long long sC,
            int K, int transA, int outBf16) {
    __shared__ __bf16 lA[128 * 32];   // [m][k]
    __shared__ __bf16 lB[128 * 32];   // [n][k]  (k-contiguous for frag loads)

    int b = blockIdx.z;
    const __bf16* Ab = A + (size_t)b * sA;
    const __bf16* Bb = B + (size_t)b * sB;

    int m_base = blockIdx.y * 128;
    int n_base = blockIdx.x * 128;

    int tid  = threadIdx.x;
    int lane = tid & 31;
    int w    = tid >> 5;
    int wm   = (w & 3) * 32;      // wave row offset within block tile
    int wn   = (w >> 2) * 64;     // wave col offset within block tile
    int lm   = lane & 15;
    int lh   = lane >> 4;         // lane half

    // loader lane mapping (fixed per thread); each thread moves packed pairs
    int pa_m = tid >> 4;          // 0..15 : row for non-trans A, step 16
    int pa_k = (tid & 15) * 2;    //        : k pair within tile
    int pt_k = tid >> 6;          // 0..3  : k for trans-A / B, step 4
    int pt_m = (tid & 63) * 2;    //        : column pair (coalesced)

    v8f acc[2][4] = {};

    for (int kt = 0; kt < K; kt += 32) {
        __syncthreads();
        // ---- stage A tile (128 x 32) bf16, layout [m][k] ----
        if (!transA) {
            const __bf16* src = Ab + (size_t)(m_base + pa_m) * ldA + (kt + pa_k);
            __bf16* dst = lA + pa_m * 32 + pa_k;
#pragma unroll 4
            for (int i = 0; i < 8; ++i)
                *(unsigned*)(dst + i * 512) =
                    *(const unsigned*)(src + (size_t)(i * 16) * ldA);
        } else {
            const __bf16* src = Ab + (size_t)(kt + pt_k) * ldA + (m_base + pt_m);
            __bf16* dst = lA + pt_m * 32 + pt_k;
#pragma unroll 4
            for (int i = 0; i < 8; ++i) {
                unsigned u = *(const unsigned*)(src + (size_t)(i * 4) * ldA);
                dst[i * 4]      = ((const __bf16*)&u)[0];
                dst[i * 4 + 32] = ((const __bf16*)&u)[1];
            }
        }
        // ---- stage B tile (32 x 128) bf16, layout [n][k] ----
        {
            const __bf16* src = Bb + (size_t)(kt + pt_k) * ldB + (n_base + pt_m);
            __bf16* dst = lB + pt_m * 32 + pt_k;
#pragma unroll 4
            for (int i = 0; i < 8; ++i) {
                unsigned u = *(const unsigned*)(src + (size_t)(i * 4) * ldB);
                dst[i * 4]      = ((const __bf16*)&u)[0];
                dst[i * 4 + 32] = ((const __bf16*)&u)[1];
            }
        }
        __syncthreads();

        // ---- A fragments: lane half kbase {0,16}; 16 contiguous k ----
        v16bf afrag[2];
#pragma unroll
        for (int i = 0; i < 2; ++i) {
            int row = (wm + i * 16 + lm) * 32;
            int kb  = lh * 16;
#pragma unroll
            for (int e = 0; e < 8; ++e) {
                afrag[i][e]     = lA[row + kb + e];        // K = kb .. kb+7
                afrag[i][e + 8] = lA[row + 8 + kb + e];    // K = kb+8 .. kb+15
            }
        }
        // ---- B fragments: double buffered, prefetch j+1 over WMMA of j ----
        v16bf bfr[2];
        {
            int col = (wn + lm) * 32 + lh * 16;
#pragma unroll
            for (int e = 0; e < 16; ++e) bfr[0][e] = lB[col + e];
        }
#pragma unroll
        for (int j = 0; j < 4; ++j) {
            if (j < 3) {
                int col = (wn + (j + 1) * 16 + lm) * 32 + lh * 16;
#pragma unroll
                for (int e = 0; e < 16; ++e) bfr[(j + 1) & 1][e] = lB[col + e];
            }
            acc[0][j] = __builtin_amdgcn_wmma_f32_16x16x32_bf16(
                false, afrag[0], false, bfr[j & 1], (short)0, acc[0][j], false, false);
            acc[1][j] = __builtin_amdgcn_wmma_f32_16x16x32_bf16(
                false, afrag[1], false, bfr[j & 1], (short)0, acc[1][j], false, false);
        }
    }

    // ---- store: C/D layout: VGPR v -> row v + 8*laneHalf, col = lane%16 ----
#pragma unroll
    for (int i = 0; i < 2; ++i) {
        int rbase = m_base + wm + i * 16 + lh * 8;
#pragma unroll
        for (int j = 0; j < 4; ++j) {
            int col = n_base + wn + j * 16 + lm;
            if (outBf16) {
                __bf16* Co = Cb + (size_t)b * sC;
#pragma unroll
                for (int v = 0; v < 8; ++v)
                    Co[(size_t)(rbase + v) * ldC + col] = (__bf16)acc[i][j][v];
            } else {
                float* Co = Cf + (size_t)b * sC;
#pragma unroll
                for (int v = 0; v < 8; ++v)
                    Co[(size_t)(rbase + v) * ldC + col] = acc[i][j][v];
            }
        }
    }
}

// ---------------------------------------------------------------------------
extern "C" void kernel_launch(void* const* d_in, const int* in_sizes, int n_in,
                              void* d_out, int out_size, void* d_ws, size_t ws_size,
                              hipStream_t stream) {
    const float* x      = (const float*)d_in[0];   // [B,N,C]
    const float* adj    = (const float*)d_in[1];   // [B,N,N]
    const int*   bnn    = (const int*)d_in[2];     // [B]
    const float* weight = (const float*)d_in[3];   // [C,1]
    const float* bias   = (const float*)d_in[4];   // [1]

    float* out    = (float*)d_out;
    float* emb    = out;                               // [B,N,C]
    float* newadj = out + (size_t)BB * NN * CC;        // [B,N,N]
    float* S      = newadj + (size_t)BB * NN * NN;     // [B,N,N]

    // f32 scratch
    float* wsf     = (float*)d_ws;
    float* xw      = wsf; wsf += BB * NN;
    float* deg_gcn = wsf; wsf += BB * NN;
    float* degA    = wsf; wsf += BB * NN;
    float* maskv   = wsf; wsf += BB * NN;
    float* alpha   = wsf; wsf += BB * NN;
    float* cut     = wsf; wsf += BB;
    // bf16 scratch (GEMM operands)
    __bf16* wsb   = (__bf16*)wsf;
    __bf16* adj16 = wsb; wsb += (size_t)BB * NN * NN;
    __bf16* S16   = wsb; wsb += (size_t)BB * NN * NN;
    __bf16* P16   = wsb; wsb += (size_t)BB * NN * NN;
    __bf16* x16   = wsb;                               // BB*NN*CC

    int rowBlocks = (BB * NN) / 8;   // 8 waves (rows) per 256-thread block

    k_rowstats <<<rowBlocks, 256, 0, stream>>>(x, adj, weight, xw, deg_gcn, degA, maskv);
    k_gcn_alpha<<<rowBlocks, 256, 0, stream>>>(adj, xw, deg_gcn, bias, alpha);
    k_select   <<<BB, NN, 0, stream>>>(alpha, bnn, cut);
    k_build_s  <<<rowBlocks, 256, 0, stream>>>(adj, alpha, degA, maskv, cut, S, S16);

    k_cvt_bf16<<<2048, 256, 0, stream>>>(adj, adj16, (long long)BB * NN * NN);
    k_cvt_bf16<<<512,  256, 0, stream>>>(x,   x16,   (long long)BB * NN * CC);

    const long long nn  = (long long)NN * NN;
    const long long ncx = (long long)NN * CC;

    // P = adj @ S  (bf16 out)
    k_wmma_gemm<<<dim3(8, 8, BB), 256, 0, stream>>>(adj16, NN, nn,
                                                    S16,   NN, nn,
                                                    nullptr, P16, NN, nn,
                                                    NN, 0, 1);
    // new_adj = S^T @ P  (f32 out)
    k_wmma_gemm<<<dim3(8, 8, BB), 256, 0, stream>>>(S16, NN, nn,
                                                    P16, NN, nn,
                                                    newadj, nullptr, NN, nn,
                                                    NN, 1, 0);
    // emb = S^T @ x  (f32 out)
    k_wmma_gemm<<<dim3(1, 8, BB), 256, 0, stream>>>(S16, NN, nn,
                                                    x16, CC, ncx,
                                                    emb, nullptr, CC, ncx,
                                                    NN, 1, 0);
}
